// TransformerEncoder_53094385713513
// MI455X (gfx1250) — matrix operand order repaired and checked by reading, not compiled, not run on hardware
//
#include <hip/hip_runtime.h>
#include <math.h>

// ---------------- problem constants ----------------
#define B_    64
#define S_    256
#define D_    768
#define H_    8
#define DH_   96
#define DFF_  2048
#define ROWS_ (B_*S_)      // 16384
#define CH_   (B_*H_)      // 512 per-head chunks
#define CHS_  (S_*DH_)     // 24576 elements per chunk
#define SD_   (S_*D_)      // 196608

// ---------------- types / helpers ----------------
typedef __attribute__((ext_vector_type(16))) __bf16 v16bf;
typedef __attribute__((ext_vector_type(8)))  float  v8f;

__device__ __forceinline__ unsigned short f2bf(float f) {
  unsigned u = __float_as_uint(f);
  u += 0x7FFFu + ((u >> 16) & 1u);        // round-to-nearest-even
  return (unsigned short)(u >> 16);
}

union FragBF { unsigned u[8]; v16bf v; };

__device__ __forceinline__ v8f wmma_bf16(const FragBF& a, const FragBF& b, v8f c) {
  // D = A(16x32 bf16) * B(32x16 bf16) + C(16x16 f32)
  return __builtin_amdgcn_wmma_f32_16x16x32_bf16(false, a.v, false, b.v,
                                                 (short)0, c, false, false);
}

// ---------------- generic WMMA GEMM ----------------
// C[M,N] = alpha * A[M,K] @ W[N,K]^T (+bias) ; optional relu / bf16-out /
// transposed store (for V^T) / split-K fp32 atomic accumulation.
// Block tile 128x128, 8 waves, wave tile 32x64 (8 WMMA per K-step), K-step 32.
template<bool OUT_BF16, bool TSTORE, bool SPLITK>
__global__ __launch_bounds__(256) void gemm_wmma(
    const unsigned short* __restrict__ A,
    const unsigned short* __restrict__ W,
    const float* __restrict__ bias,
    float* __restrict__ Cf,
    unsigned short* __restrict__ Cb,
    int M, int N, int K,
    int lda, int ldw, int ldc,
    long long sA, long long sW, long long sC,
    float alpha, int relu, int kPerSplit)
{
  __shared__ unsigned ldsA[128 * 17];   // 128 rows x 32 bf16 (stride 17 dwords)
  __shared__ unsigned ldsB[128 * 17];   // 128 rows x 32 bf16

  const int t    = threadIdx.x;
  const int lane = t & 31;
  const int l15  = lane & 15;
  const int half = lane >> 4;
  const int w    = t >> 5;
  const int wm   = w & 3;     // wave M offset = wm*32
  const int wn   = w >> 2;    // wave N offset = wn*64

  const int m0 = blockIdx.y * 128;
  const int n0 = blockIdx.x * 128;

  const unsigned short* Ab;
  const unsigned short* Wb;
  long long coff;
  int k0, kend;
  if constexpr (SPLITK) {
    Ab = A; Wb = W; coff = 0;
    k0 = blockIdx.z * kPerSplit; kend = k0 + kPerSplit;
  } else {
    Ab = A + (long long)blockIdx.z * sA;
    Wb = W + (long long)blockIdx.z * sW;
    coff = (long long)blockIdx.z * sC;
    k0 = 0; kend = K;
  }

  v8f acc[2][4] = {};

  const int sr   = t >> 1;          // staging row 0..127 (A and B)
  const int scol = (t & 1) << 4;    // 0 / 16 (bf16 elements)

  for (int kk = k0; kk < kend; kk += 32) {
    // ---- stage A tile (128 x 32 bf16), zero-padded past M ----
    {
      uint4 v0 = {0,0,0,0}, v1 = {0,0,0,0};
      if (m0 + sr < M) {
        const uint4* p = reinterpret_cast<const uint4*>(
            Ab + (long long)(m0 + sr) * lda + kk + scol);
        v0 = p[0]; v1 = p[1];
      }
      unsigned* dst = &ldsA[sr * 17 + (scol >> 1)];
      dst[0]=v0.x; dst[1]=v0.y; dst[2]=v0.z; dst[3]=v0.w;
      dst[4]=v1.x; dst[5]=v1.y; dst[6]=v1.z; dst[7]=v1.w;
    }
    // ---- stage B tile (128 x 32 bf16), zero-padded past N ----
    {
      uint4 v0 = {0,0,0,0}, v1 = {0,0,0,0};
      if (n0 + sr < N) {
        const uint4* p = reinterpret_cast<const uint4*>(
            Wb + (long long)(n0 + sr) * ldw + kk + scol);
        v0 = p[0]; v1 = p[1];
      }
      unsigned* dst = &ldsB[sr * 17 + (scol >> 1)];
      dst[0]=v0.x; dst[1]=v0.y; dst[2]=v0.z; dst[3]=v0.w;
      dst[4]=v1.x; dst[5]=v1.y; dst[6]=v1.z; dst[7]=v1.w;
    }
    if (kk + 32 < kend && m0 + sr < M) {   // global_prefetch_b8 next K chunk
      __builtin_prefetch(Ab + (long long)(m0 + sr) * lda + kk + 32 + scol, 0, 0);
    }
    __syncthreads();

    // ---- fragment gathers per documented wave32 layouts ----
    FragBF a[2], b[4];
    #pragma unroll
    for (int fi = 0; fi < 2; ++fi) {
      const unsigned* p = &ldsA[(wm*32 + fi*16 + l15) * 17];
      #pragma unroll
      for (int j = 0; j < 4; ++j) {
        a[fi].u[j]   = p[half*4 + j];
        a[fi].u[4+j] = p[8 + half*4 + j];
      }
    }
    #pragma unroll
    for (int fj = 0; fj < 4; ++fj) {
      const unsigned* r = &ldsB[(wn*64 + fj*16 + l15) * 17];
      #pragma unroll
      for (int j = 0; j < 4; ++j) {
        b[fj].u[j]   = r[half*4 + j];
        b[fj].u[4+j] = r[8 + half*4 + j];
      }
    }
    #pragma unroll
    for (int fi = 0; fi < 2; ++fi)
    #pragma unroll
    for (int fj = 0; fj < 4; ++fj)
      acc[fi][fj] = wmma_bf16(a[fi], b[fj], acc[fi][fj]);
    __syncthreads();
  }

  // ---- epilogue: C frag element r -> (m = r + 8*half, n = l15) ----
  #pragma unroll
  for (int fi = 0; fi < 2; ++fi)
  #pragma unroll
  for (int fj = 0; fj < 4; ++fj)
  #pragma unroll
  for (int r = 0; r < 8; ++r) {
    int mg = m0 + wm*32 + fi*16 + r + half*8;
    int ng = n0 + wn*64 + fj*16 + l15;
    if (mg < M && ng < N) {
      float v = acc[fi][fj][r] * alpha;
      if (bias) v += bias[ng];
      if (relu) v = fmaxf(v, 0.f);
      if constexpr (OUT_BF16) {
        if constexpr (TSTORE) Cb[coff + (long long)ng * ldc + mg] = f2bf(v);
        else                  Cb[coff + (long long)mg * ldc + ng] = f2bf(v);
      } else {
        if constexpr (SPLITK) atomicAdd(&Cf[(long long)mg * ldc + ng], v);
        else                  Cf[coff + (long long)mg * ldc + ng] = v;
      }
    }
  }
}

// ---------------- softmax over 256-wide rows ----------------
__global__ __launch_bounds__(256) void softmax_rows(
    const float* __restrict__ Sc, unsigned short* __restrict__ P)
{
  const int t = threadIdx.x;
  const long long base = (long long)blockIdx.x * 256;
  __shared__ float red[256];
  float v = Sc[base + t];
  red[t] = v;
  for (int st = 128; st > 0; st >>= 1) { __syncthreads(); if (t < st) red[t] = fmaxf(red[t], red[t+st]); }
  __syncthreads();
  float mx = red[0];
  __syncthreads();
  float e = __expf(v - mx);
  red[t] = e;
  for (int st = 128; st > 0; st >>= 1) { __syncthreads(); if (t < st) red[t] += red[t+st]; }
  __syncthreads();
  P[base + t] = f2bf(e * (1.f / red[0]));
}

// ---------------- residual add + LayerNorm (768), dual f32/bf16 write ----------------
__global__ __launch_bounds__(256) void add_layernorm(
    const float* __restrict__ xin, const float* __restrict__ dl,
    const float* __restrict__ g, const float* __restrict__ b,
    float* __restrict__ xout, unsigned short* __restrict__ xb)
{
  const int t = threadIdx.x;
  const long long base = (long long)blockIdx.x * D_;
  __shared__ float s1[256], s2[256];
  float v[3]; float a = 0.f, q = 0.f;
  #pragma unroll
  for (int i = 0; i < 3; ++i) {
    int c = t + i*256;
    float x = xin[base + c] + dl[base + c];
    v[i] = x; a += x; q += x*x;
  }
  s1[t] = a; s2[t] = q;
  for (int st = 128; st > 0; st >>= 1) { __syncthreads(); if (t < st) { s1[t]+=s1[t+st]; s2[t]+=s2[t+st]; } }
  __syncthreads();
  float mean = s1[0] * (1.f / D_);
  float var  = s2[0] * (1.f / D_) - mean*mean;
  float rstd = rsqrtf(var + 1e-5f);
  #pragma unroll
  for (int i = 0; i < 3; ++i) {
    int c = t + i*256;
    float y = (v[i] - mean) * rstd * g[c] + b[c];
    xout[base + c] = y; xb[base + c] = f2bf(y);
  }
}

// ---------------- LayerNorm over 196608 + ReLU6 -> bf16 ----------------
__global__ __launch_bounds__(1024) void big_ln_clip(
    const float* __restrict__ x, const float* __restrict__ g,
    const float* __restrict__ b, unsigned short* __restrict__ out)
{
  const int t = threadIdx.x;
  const long long base = (long long)blockIdx.x * SD_;
  __shared__ float s1[1024], s2[1024];
  float a = 0.f, q = 0.f;
  for (int c = t; c < SD_; c += 1024) { float v = x[base + c]; a += v; q += v*v; }
  s1[t] = a; s2[t] = q;
  for (int st = 512; st > 0; st >>= 1) { __syncthreads(); if (t < st) { s1[t]+=s1[t+st]; s2[t]+=s2[t+st]; } }
  __syncthreads();
  float mean = s1[0] * (1.f / SD_);
  float var  = s2[0] * (1.f / SD_) - mean*mean;
  float rstd = rsqrtf(var + 1e-5f);
  for (int c = t; c < SD_; c += 1024) {
    float y = (x[base + c] - mean) * rstd * g[c] + b[c];
    y = fminf(fmaxf(y, 0.f), 6.f);
    out[base + c] = f2bf(y);
  }
}

// ---------------- bias + LayerNorm(768) + ReLU6 -> f32 ----------------
__global__ __launch_bounds__(256) void ln_clip_768(
    const float* __restrict__ xin, const float* __restrict__ bias,
    const float* __restrict__ g, const float* __restrict__ b,
    float* __restrict__ out)
{
  const int t = threadIdx.x;
  const long long base = (long long)blockIdx.x * D_;
  __shared__ float s1[256], s2[256];
  float v[3]; float a = 0.f, q = 0.f;
  #pragma unroll
  for (int i = 0; i < 3; ++i) {
    int c = t + i*256;
    float x = xin[base + c] + bias[c];
    v[i] = x; a += x; q += x*x;
  }
  s1[t] = a; s2[t] = q;
  for (int st = 128; st > 0; st >>= 1) { __syncthreads(); if (t < st) { s1[t]+=s1[t+st]; s2[t]+=s2[t+st]; } }
  __syncthreads();
  float mean = s1[0] * (1.f / D_);
  float var  = s2[0] * (1.f / D_) - mean*mean;
  float rstd = rsqrtf(var + 1e-5f);
  #pragma unroll
  for (int i = 0; i < 3; ++i) {
    int c = t + i*256;
    float y = (v[i] - mean) * rstd * g[c] + b[c];
    out[base + c] = fminf(fmaxf(y, 0.f), 6.f);
  }
}

// ---------------- final (64,768) @ (2,768)^T + bias ----------------
__global__ __launch_bounds__(256) void final_head(
    const float* __restrict__ f, const float* __restrict__ w,
    const float* __restrict__ bw, float* __restrict__ out)
{
  const int t = threadIdx.x;
  const int bb = blockIdx.x >> 1, o = blockIdx.x & 1;
  __shared__ float s[256];
  float a = 0.f;
  for (int c = t; c < D_; c += 256)
    a += f[(long long)bb * D_ + c] * w[(long long)o * D_ + c];
  s[t] = a;
  for (int st = 128; st > 0; st >>= 1) { __syncthreads(); if (t < st) s[t] += s[t+st]; }
  __syncthreads();
  if (t == 0) out[bb*2 + o] = s[0] + bw[o];
}

// ---------------- embedding + positional encoding + mask ----------------
__global__ __launch_bounds__(256) void embed_pos(
    const int* __restrict__ tokens, const float* __restrict__ masks,
    const float* __restrict__ emb, float* __restrict__ x,
    unsigned short* __restrict__ xb)
{
  const int row = blockIdx.x;          // b*256 + s
  const int t = threadIdx.x;
  const int s = row & (S_ - 1);
  const int tok = tokens[row];
  const float mk = masks[row];
  float fr = powf(10000.f, (2.f * floorf((float)s * 0.5f)) * (1.f / (float)D_));
  float bbv = (float)s / fr;
  float sn = sinf(bbv), cs = cosf(bbv);
  const long long base = (long long)row * D_;
  const long long eb = (long long)tok * D_;
  #pragma unroll
  for (int i = 0; i < 3; ++i) {
    int c = t + i*256;
    float v = emb[eb + c] + (((c & 1) == 0) ? sn : cs);
    v *= mk;
    x[base + c] = v; xb[base + c] = f2bf(v);
  }
}

// ---------------- fp32 -> bf16 weight conversion ----------------
__global__ __launch_bounds__(256) void cvt_bf16(
    const float* __restrict__ in, unsigned short* __restrict__ out, long long n)
{
  long long i = (long long)blockIdx.x * 256 + threadIdx.x;
  long long stride = (long long)gridDim.x * 256;
  for (; i < n; i += stride) out[i] = f2bf(in[i]);
}

// ---------------- host orchestration ----------------
extern "C" void kernel_launch(void* const* d_in, const int* in_sizes, int n_in,
                              void* d_out, int out_size, void* d_ws, size_t ws_size,
                              hipStream_t stream)
{
  (void)in_sizes; (void)n_in; (void)out_size; (void)ws_size;
  const int*   tokens = (const int*)  d_in[0];
  const float* masks  = (const float*)d_in[1];
  const float* emb    = (const float*)d_in[2];
  const float* wq  = (const float*)d_in[3];  const float* bq  = (const float*)d_in[4];
  const float* wk  = (const float*)d_in[5];  const float* bk  = (const float*)d_in[6];
  const float* wv  = (const float*)d_in[7];  const float* bv  = (const float*)d_in[8];
  const float* wo  = (const float*)d_in[9];  const float* bo  = (const float*)d_in[10];
  const float* lng = (const float*)d_in[11]; const float* lnb = (const float*)d_in[12];
  const float* w1  = (const float*)d_in[13]; const float* fb1b= (const float*)d_in[14];
  const float* w2  = (const float*)d_in[15]; const float* fb2b= (const float*)d_in[16];
  const float* fg1 = (const float*)d_in[17]; const float* fbb1= (const float*)d_in[18];
  const float* fw1 = (const float*)d_in[19]; const float* fbw1= (const float*)d_in[20];
  const float* fg2 = (const float*)d_in[21]; const float* fbb2= (const float*)d_in[22];
  const float* fw2 = (const float*)d_in[23]; const float* fbw2= (const float*)d_in[24];

  // ---- workspace carve (256B aligned) ----
  char* wsp = (char*)d_ws; size_t off = 0;
  auto take = [&](size_t bytes) -> void* {
    void* r = wsp + off; off = (off + bytes + 255) & ~(size_t)255; return r;
  };
  float*          xf   = (float*)         take((size_t)ROWS_*D_*4);
  float*          tmp  = (float*)         take((size_t)ROWS_*D_*4);
  unsigned short* xb   = (unsigned short*)take((size_t)ROWS_*D_*2);
  unsigned short* qb   = (unsigned short*)take((size_t)ROWS_*D_*2);
  unsigned short* kb   = (unsigned short*)take((size_t)ROWS_*D_*2);
  unsigned short* vTb  = (unsigned short*)take((size_t)ROWS_*D_*2);
  unsigned short* ob   = (unsigned short*)take((size_t)ROWS_*D_*2);
  float*          sc   = (float*)         take((size_t)CH_*S_*S_*4);
  unsigned short* Pb   = (unsigned short*)take((size_t)CH_*S_*S_*2);
  unsigned short* ff1  = (unsigned short*)take((size_t)ROWS_*DFF_*2);
  unsigned short* fbuf = (unsigned short*)take((size_t)B_*SD_*2);
  unsigned short* wq_b = (unsigned short*)take((size_t)DH_*DH_*2);
  unsigned short* wk_b = (unsigned short*)take((size_t)DH_*DH_*2);
  unsigned short* wv_b = (unsigned short*)take((size_t)DH_*DH_*2);
  unsigned short* wo_b = (unsigned short*)take((size_t)D_*D_*2);
  unsigned short* w1_b = (unsigned short*)take((size_t)DFF_*D_*2);
  unsigned short* w2_b = (unsigned short*)take((size_t)D_*DFF_*2);
  unsigned short* fw1b = (unsigned short*)take((size_t)D_*SD_*2);
  float*          g1   = (float*)         take((size_t)B_*D_*4);
  float*          g2   = (float*)         take((size_t)B_*D_*4);

  auto cvt = [&](const float* in, unsigned short* o, long long n) {
    int blocks = (int)((n + 2047) / 2048);
    if (blocks > 16384) blocks = 16384;
    if (blocks < 1) blocks = 1;
    cvt_bf16<<<blocks, 256, 0, stream>>>(in, o, n);
  };
  cvt(wq, wq_b, (long long)DH_*DH_);
  cvt(wk, wk_b, (long long)DH_*DH_);
  cvt(wv, wv_b, (long long)DH_*DH_);
  cvt(wo, wo_b, (long long)D_*D_);
  cvt(w1, w1_b, (long long)DFF_*D_);
  cvt(w2, w2_b, (long long)D_*DFF_);
  cvt(fw1, fw1b, (long long)D_*SD_);

  embed_pos<<<ROWS_, 256, 0, stream>>>(tokens, masks, emb, xf, xb);

  const float scale = 0.10206207261596577f;  // 1/sqrt(96)
  const long long SS = (long long)S_ * S_;

  for (int layer = 0; layer < 6; ++layer) {
    // per-head QKV: 512 contiguous (256x96) chunks @ (96x96)^T
    gemm_wmma<true,false,false><<<dim3(1,2,CH_),256,0,stream>>>(
        xb, wq_b, bq, nullptr, qb, S_, DH_, DH_, DH_, DH_, DH_,
        CHS_, 0, CHS_, 1.f, 0, 0);
    gemm_wmma<true,false,false><<<dim3(1,2,CH_),256,0,stream>>>(
        xb, wk_b, bk, nullptr, kb, S_, DH_, DH_, DH_, DH_, DH_,
        CHS_, 0, CHS_, 1.f, 0, 0);
    gemm_wmma<true,true,false><<<dim3(1,2,CH_),256,0,stream>>>(   // V stored transposed
        xb, wv_b, bv, nullptr, vTb, S_, DH_, DH_, DH_, DH_, S_,
        CHS_, 0, CHS_, 1.f, 0, 0);
    // scores = scale * Q @ K^T   (W = K activations as (N=256,K=96))
    gemm_wmma<false,false,false><<<dim3(2,2,CH_),256,0,stream>>>(
        qb, kb, nullptr, sc, nullptr, S_, S_, DH_, DH_, DH_, S_,
        CHS_, CHS_, SS, scale, 0, 0);
    softmax_rows<<<CH_*S_, 256, 0, stream>>>(sc, Pb);
    // O = P @ V  (W = V^T as (N=96,K=256))
    gemm_wmma<true,false,false><<<dim3(1,2,CH_),256,0,stream>>>(
        Pb, vTb, nullptr, nullptr, ob, S_, DH_, S_, S_, S_, DH_,
        SS, CHS_, CHS_, 1.f, 0, 0);
    // output projection (16384,768)@(768,768)^T
    gemm_wmma<false,false,false><<<dim3(6,128,1),256,0,stream>>>(
        ob, wo_b, bo, tmp, nullptr, ROWS_, D_, D_, D_, D_, D_,
        0, 0, 0, 1.f, 0, 0);
    add_layernorm<<<ROWS_, 256, 0, stream>>>(xf, tmp, lng, lnb, xf, xb);
    // FFN1 (relu, bf16 out) and FFN2
    gemm_wmma<true,false,false><<<dim3(16,128,1),256,0,stream>>>(
        xb, w1_b, fb1b, nullptr, ff1, ROWS_, DFF_, D_, D_, D_, DFF_,
        0, 0, 0, 1.f, 1, 0);
    gemm_wmma<false,false,false><<<dim3(6,128,1),256,0,stream>>>(
        ff1, w2_b, fb2b, tmp, nullptr, ROWS_, D_, DFF_, DFF_, DFF_, D_,
        0, 0, 0, 1.f, 0, 0);
    add_layernorm<<<ROWS_, 256, 0, stream>>>(xf, tmp, lng, lnb, xf, xb);
  }

  // final head
  big_ln_clip<<<B_, 1024, 0, stream>>>(xf, fg1, fbb1, fbuf);
  hipMemsetAsync(g1, 0, (size_t)B_*D_*4, stream);
  // (64,196608) @ (768,196608)^T with 96-way split-K (bandwidth-bound, fills device)
  gemm_wmma<false,false,true><<<dim3(6,1,96),256,0,stream>>>(
      fbuf, fw1b, nullptr, g1, nullptr, B_, D_, SD_, SD_, SD_, D_,
      0, 0, 0, 1.f, 0, SD_/96);
  ln_clip_768<<<B_, 256, 0, stream>>>(g1, fbw1, fg2, fbb2, g2);
  final_head<<<B_*2, 256, 0, stream>>>(g2, fw2, fbw2, (float*)d_out);
}
